// MentalPolicy_55207509623093
// MI455X (gfx1250) — compile-verified, matrix operand order checked by reading, not run on hardware
//
#include <hip/hip_runtime.h>
#include <cstddef>

// ---------------------------------------------------------------------------
// MI455X / gfx1250 implementation.
//   T=65536, DIM_S=64, DIM_A=8, DIM_C=16, H=256
// Phase 0: pack fp32 weights -> f16 in WMMA B-fragment lane order (workspace)
// Phase 1: fused dual-MLP kernel, 1 wave / 16 rows, V_WMMA_F32_16X16X32_F16,
//          s-tile staged via TDM tensor_load_to_lds (TENSORcnt)
//          -> writes log_pis, log_trs, entropy, log_tr0(ws)
// Phase 2: chunked log-semiring parallel scan for log_alpha / log_beta,
//          with global_prefetch of the next step matrix
// ---------------------------------------------------------------------------

typedef __attribute__((ext_vector_type(16))) _Float16 v16h;
typedef __attribute__((ext_vector_type(8)))  float    v8f;
typedef __attribute__((ext_vector_type(4)))  unsigned int u32x4;
typedef __attribute__((ext_vector_type(8)))  int      i32x8;
typedef __attribute__((ext_vector_type(4)))  int      i32x4;

#define TT        65536
#define KMAX      65535      /* T-1 = number of N_k step matrices  */
#define NCH       256        /* scan chunks                        */
#define CHL       256        /* chunk length                       */
#define HALF_LOG2PI 0.91893853320467274f

#if defined(__has_builtin)
#if __has_builtin(__builtin_amdgcn_tensor_load_to_lds) && \
    __has_builtin(__builtin_amdgcn_s_wait_tensorcnt)
#define HAVE_TDM 1
#endif
#endif
#ifndef HAVE_TDM
#define HAVE_TDM 0
#endif

// ---------------------------------------------------------------------------
// Phase 0: weight packing.  B-fragment layout (mirror of ISA 16-bit A layout):
// lane l holds column n = nt*16 + (l&15); its v16h element e maps to
//   k = 32*kt + ((e<8) ? 8*h + e : 16 + 8*h + (e-8)),  h = l>>4
// so each lane's 16 halves are two contiguous 8-half runs -> b128 loads.
// ---------------------------------------------------------------------------
__global__ __launch_bounds__(256) void pack_w(const float* __restrict__ src,
                                              _Float16* __restrict__ dst,
                                              int K, int N, int total) {
  int id = blockIdx.x * blockDim.x + threadIdx.x;
  if (id >= total) return;
  int e    = id & 15;
  int l    = (id >> 4) & 31;
  int tile = id >> 9;
  int KT   = K >> 5;
  int kt   = tile % KT;
  int nt   = tile / KT;
  int h    = l >> 4;
  int k    = kt * 32 + ((e < 8) ? (8 * h + e) : (16 + 8 * h + (e - 8)));
  int n    = nt * 16 + (l & 15);
  dst[id]  = (_Float16)src[(size_t)k * N + n];
}

// ---------------------------------------------------------------------------
// TDM: DMA a 16x64 fp32 tile (row-major, stride 64) from global into LDS.
// D# per CDNA5 ISA ch.8: group0 = {flags, lds_addr, global_addr, type=2},
// group1 = {data_size=4B, tensor 64x16, tile 64x16, dim0_stride=64},
// groups 2/3 zero (2-D tensor).  LDS offset = low 32 bits of flat address
// ---------------------------------------------------------------------------
#if HAVE_TDM
__device__ __forceinline__ void tdm_load_s_tile(const float* gptr, void* ldsptr) {
  unsigned lds = (unsigned)(unsigned long long)ldsptr;
  unsigned long long ga = (unsigned long long)gptr;
  u32x4 g0 = { 1u,                                   // count=1, user D#
               lds,                                  // lds_addr (bytes)
               (unsigned)(ga & 0xffffffffu),         // global_addr[31:0]
               (unsigned)((ga >> 32) & 0x01ffffffu)  // global_addr[56:32]
                 | (2u << 30) };                     // type = 2 ("image")
  i32x8 g1 = { (int)(2u << 16),        // data_size=2 -> 4 bytes
               (int)(64u << 16),       // tensor_dim0 = 64 (bits 79:48 lo)
               (int)(16u << 16),       // tensor_dim1 = 16 (bits 95:80)
               (int)(64u << 16),       // tile_dim0 = 64 (bits 127:112)
               16,                     // tile_dim1 = 16 (bits 143:128)
               64,                     // tensor_dim0_stride = 64
               0, 0 };
  i32x4 gz = { 0, 0, 0, 0 };
#if __clang_major__ >= 23
  i32x8 gz8 = { 0, 0, 0, 0, 0, 0, 0, 0 };
  __builtin_amdgcn_tensor_load_to_lds(g0, g1, gz, gz, gz8, 0);
#else
  __builtin_amdgcn_tensor_load_to_lds(g0, g1, gz, gz, 0);
#endif
}
#endif

// ---------------------------------------------------------------------------
// WMMA fragment helpers (A from LDS activations, B from packed global)
// ---------------------------------------------------------------------------
template <int STRIDE>
__device__ __forceinline__ v16h lds_afrag(const _Float16 (&act)[16][STRIDE],
                                          int m, int kt, int h) {
  union { v16h v; uint4 q[2]; } u;
  u.q[0] = *(const uint4*)&act[m][kt * 32 + 8 * h];
  u.q[1] = *(const uint4*)&act[m][kt * 32 + 16 + 8 * h];
  return u.v;
}

__device__ __forceinline__ v16h g_bfrag(const _Float16* __restrict__ pb,
                                        int nt, int KT, int kt, int l) {
  union { v16h v; uint4 q[2]; } u;
  const uint4* p = (const uint4*)(pb + (((size_t)nt * KT + kt) * 32 + l) * 16);
  u.q[0] = p[0];
  u.q[1] = p[1];
  return u.v;
}

template <int K, int STRIDE>
__device__ __forceinline__ v8f tile_acc(const _Float16 (&act)[16][STRIDE],
                                        const _Float16* __restrict__ pb,
                                        int nt, int l) {
  const int KT = K / 32;
  const int h = l >> 4, m = l & 15;
  v8f c = {};
#pragma unroll
  for (int kt = 0; kt < KT; ++kt) {
    v16h a = lds_afrag<STRIDE>(act, m, kt, h);
    v16h b = g_bfrag(pb, nt, KT, kt, l);
    c = __builtin_amdgcn_wmma_f32_16x16x32_f16(false, a, false, b,
                                               (short)0, c, false, false);
  }
  return c;
}

// ---------------------------------------------------------------------------
// Phase 1: fused dual-MLP.  grid = T/16 blocks, 32 threads (one wave,
// EXEC all-ones as WMMA requires).  LDS ops within a wave are in-order,
// so no barriers are needed.
// ---------------------------------------------------------------------------
__global__ __launch_bounds__(32) void mlp_kernel(
    const float* __restrict__ s, const float* __restrict__ a_arr,
    const _Float16* __restrict__ w1p, const _Float16* __restrict__ w2p,
    const _Float16* __restrict__ w3p, const _Float16* __restrict__ w1o,
    const _Float16* __restrict__ w2o, const _Float16* __restrict__ w3o,
    const float* __restrict__ pb1, const float* __restrict__ pb2,
    const float* __restrict__ pb3, const float* __restrict__ ob1,
    const float* __restrict__ ob2, const float* __restrict__ ob3,
    const float* __restrict__ a_log_std,
    float* __restrict__ o_lpis, float* __restrict__ o_ltrs,
    float* __restrict__ o_ent, float* __restrict__ ws_logtr0) {
  __shared__ _Float16 sA[16][64];     // state tile, f16
  __shared__ _Float16 act1[16][256];  // h1 (ping)
  __shared__ _Float16 act2[16][256];  // h2 (pong)
  __shared__ float    fbuf[16][272];  // fp32 staging: mean(128) then logits(272)

  const int l   = threadIdx.x;
  const int t0  = blockIdx.x * 16;
  const int h   = l >> 4;
  const int nlo = l & 15;

#if HAVE_TDM
  // TDM DMA the fp32 tile into LDS, then convert to f16 in LDS.
  __shared__ float sF32[16][64];
  tdm_load_s_tile(s + (size_t)t0 * 64, (void*)&sF32[0][0]);
  __builtin_amdgcn_s_wait_tensorcnt(0);
  for (int idx = l; idx < 16 * 64; idx += 32)
    sA[idx >> 6][idx & 63] = (_Float16)sF32[idx >> 6][idx & 63];
#else
  for (int idx = l; idx < 16 * 64; idx += 32)
    sA[idx >> 6][idx & 63] = (_Float16)s[(size_t)(t0 + (idx >> 6)) * 64 + (idx & 63)];
#endif

  // ---------------- policy MLP ----------------
  for (int nt = 0; nt < 16; ++nt) {               // L1: 64 -> 256, relu
    v8f c = tile_acc<64>(sA, w1p, nt, l);
    int n = nt * 16 + nlo; float bn = pb1[n];
#pragma unroll
    for (int r = 0; r < 8; ++r) {
      float v = c[r] + bn;
      act1[8 * h + r][n] = (_Float16)(v > 0.f ? v : 0.f);
    }
  }
  for (int nt = 0; nt < 16; ++nt) {               // L2: 256 -> 256, relu
    v8f c = tile_acc<256>(act1, w2p, nt, l);
    int n = nt * 16 + nlo; float bn = pb2[n];
#pragma unroll
    for (int r = 0; r < 8; ++r) {
      float v = c[r] + bn;
      act2[8 * h + r][n] = (_Float16)(v > 0.f ? v : 0.f);
    }
  }
  for (int nt = 0; nt < 8; ++nt) {                // L3: 256 -> 128, clip
    v8f c = tile_acc<256>(act2, w3p, nt, l);
    int n = nt * 16 + nlo; float bn = pb3[n];
#pragma unroll
    for (int r = 0; r < 8; ++r) {
      float v = c[r] + bn;
      v = v > 10.f ? 10.f : (v < -10.f ? -10.f : v);
      fbuf[8 * h + r][n] = v;
    }
  }

  // ---------------- Gaussian log-probs ----------------
  float lstd[8], istd[8];
#pragma unroll
  for (int d = 0; d < 8; ++d) {
    float x  = tanhf(a_log_std[d]);
    float ls = -5.f + 3.5f * (x + 1.f);
    lstd[d]  = ls;
    istd[d]  = expf(-ls);
  }
  for (int p = l; p < 256; p += 32) {             // 16 rows x 16 options
    int row = p >> 4, cc = p & 15;
    float acc = -8.f * HALF_LOG2PI;
#pragma unroll
    for (int d = 0; d < 8; ++d) {
      float at = a_arr[(size_t)(t0 + row) * 8 + d];
      float z  = (at - fbuf[row][cc * 8 + d]) * istd[d];
      acc += -0.5f * z * z - lstd[d];
    }
    o_lpis[(size_t)(t0 + row) * 16 + cc] = acc;
  }

  // ---------------- option MLP ----------------
  for (int nt = 0; nt < 16; ++nt) {               // L1
    v8f c = tile_acc<64>(sA, w1o, nt, l);
    int n = nt * 16 + nlo; float bn = ob1[n];
#pragma unroll
    for (int r = 0; r < 8; ++r) {
      float v = c[r] + bn;
      act1[8 * h + r][n] = (_Float16)(v > 0.f ? v : 0.f);
    }
  }
  for (int nt = 0; nt < 16; ++nt) {               // L2
    v8f c = tile_acc<256>(act1, w2o, nt, l);
    int n = nt * 16 + nlo; float bn = ob2[n];
#pragma unroll
    for (int r = 0; r < 8; ++r) {
      float v = c[r] + bn;
      act2[8 * h + r][n] = (_Float16)(v > 0.f ? v : 0.f);
    }
  }
  for (int nt = 0; nt < 17; ++nt) {               // L3: 256 -> 272
    v8f c = tile_acc<256>(act2, w3o, nt, l);
    int n = nt * 16 + nlo; float bn = ob3[n];
#pragma unroll
    for (int r = 0; r < 8; ++r)
      fbuf[8 * h + r][n] = c[r] + bn;
  }

  // ---------------- log_softmax over 17 groups of 16 ----------------
  for (int g = l; g < 16 * 17; g += 32) {
    int row = g / 17, comp = g % 17;
    long t = (long)t0 + row;
    float mx = -1e30f;
#pragma unroll
    for (int j = 0; j < 16; ++j) {
      float v = fbuf[row][comp * 16 + j];
      mx = v > mx ? v : mx;
    }
    float se = 0.f;
#pragma unroll
    for (int j = 0; j < 16; ++j) se += expf(fbuf[row][comp * 16 + j] - mx);
    float lse = mx + logf(se);
    if (comp < 16 && t >= 1) {
      float ent = 0.f;
      size_t base = ((size_t)(t - 1) * 16 + comp) * 16;
#pragma unroll
      for (int j = 0; j < 16; ++j) {
        float v = fbuf[row][comp * 16 + j] - lse;
        o_ltrs[base + j] = v;
        ent -= v * expf(v);
      }
      o_ent[(size_t)(t - 1) * 16 + comp] = ent;
    } else if (comp == 16 && t == 0) {
#pragma unroll
      for (int j = 0; j < 16; ++j)
        ws_logtr0[j] = fbuf[row][16 * 16 + j] - lse;
    }
  }
}

// ---------------------------------------------------------------------------
// Phase 2a: per-chunk log-semiring matrix product.
// N_k[i][j] = log_trs[k][i][j] + log_pis[k+1][j];  M_c = N_{cL} * ... (lse-matmul)
// ---------------------------------------------------------------------------
__global__ __launch_bounds__(256) void scan_chunks(const float* __restrict__ ltrs,
                                                   const float* __restrict__ lpis,
                                                   float* __restrict__ Mout) {
  __shared__ float M[16][16];
  __shared__ float Nk[16][16];
  const int tid = threadIdx.x;
  const int i = tid >> 4, j = tid & 15;
  const int k0 = blockIdx.x * CHL;
  const int k1 = (k0 + CHL < KMAX) ? k0 + CHL : KMAX;
  M[i][j] = (i == j) ? 0.f : -1e30f;
  __syncthreads();
  for (int k = k0; k < k1; ++k) {
    Nk[i][j] = ltrs[(size_t)k * 256 + tid] + lpis[(size_t)(k + 1) * 16 + j];
    if (k + 1 < k1)   // hide L2 latency of the next step matrix
      __builtin_prefetch(ltrs + (size_t)(k + 1) * 256 + tid, 0, 1);
    __syncthreads();
    float mx = -1e30f;
#pragma unroll
    for (int q = 0; q < 16; ++q) {
      float v = M[i][q] + Nk[q][j];
      mx = v > mx ? v : mx;
    }
    float s = 0.f;
#pragma unroll
    for (int q = 0; q < 16; ++q) s += expf(M[i][q] + Nk[q][j] - mx);
    float out = mx + logf(s);
    __syncthreads();
    M[i][j] = out;
  }
  __syncthreads();
  Mout[(size_t)blockIdx.x * 256 + tid] = M[i][j];
}

// ---------------------------------------------------------------------------
// Phase 2b: chunk-boundary vectors.  Lanes 0-15 sweep forward (alpha),
// lanes 16-31 sweep backward (beta), concurrently.
// ---------------------------------------------------------------------------
__global__ __launch_bounds__(32) void scan_bounds(const float* __restrict__ Mw,
                                                  const float* __restrict__ logtr0,
                                                  const float* __restrict__ lpis,
                                                  float* __restrict__ ab,
                                                  float* __restrict__ bb,
                                                  float* __restrict__ o_alpha,
                                                  float* __restrict__ o_beta) {
  __shared__ float av[16], bv[16], ta[16], tb[16];
  const int l = threadIdx.x;
  if (l < 16) {
    float a0 = logtr0[l] + lpis[l];
    av[l] = a0; bv[l] = 0.f;
    ab[l] = a0; bb[(size_t)NCH * 16 + l] = 0.f;
    o_alpha[l] = a0;
    o_beta[(size_t)(TT - 1) * 16 + l] = 0.f;
  }
  __syncthreads();
  for (int c = 0; c < NCH; ++c) {
    if (l < 16) {                 // forward: av = av (row-vec) * M_c
      const float* M = Mw + (size_t)c * 256;
      float mx = -1e30f;
#pragma unroll
      for (int q = 0; q < 16; ++q) { float v = av[q] + M[q * 16 + l]; mx = v > mx ? v : mx; }
      float s = 0.f;
#pragma unroll
      for (int q = 0; q < 16; ++q) s += expf(av[q] + M[q * 16 + l] - mx);
      ta[l] = mx + logf(s);
    } else {                      // backward: bv = M_{NCH-1-c} * bv (col-vec)
      const int i = l - 16;
      const float* M = Mw + (size_t)(NCH - 1 - c) * 256;
      float mx = -1e30f;
#pragma unroll
      for (int q = 0; q < 16; ++q) { float v = M[i * 16 + q] + bv[q]; mx = v > mx ? v : mx; }
      float s = 0.f;
#pragma unroll
      for (int q = 0; q < 16; ++q) s += expf(M[i * 16 + q] + bv[q] - mx);
      tb[i] = mx + logf(s);
    }
    __syncthreads();
    if (l < 16) { av[l] = ta[l]; ab[(size_t)(c + 1) * 16 + l] = ta[l]; }
    else        { int i = l - 16; bv[i] = tb[i]; bb[(size_t)(NCH - 1 - c) * 16 + i] = tb[i]; }
    __syncthreads();
  }
}

// ---------------------------------------------------------------------------
// Phase 2c: intra-chunk replay (mat-vec per step).  Blocks 0..255: forward
// (alpha rows cL+1..end); blocks 256..511: backward (beta rows cL..end-1).
// ---------------------------------------------------------------------------
__global__ __launch_bounds__(32) void scan_fill(const float* __restrict__ ltrs,
                                                const float* __restrict__ lpis,
                                                const float* __restrict__ ab,
                                                const float* __restrict__ bb,
                                                float* __restrict__ o_alpha,
                                                float* __restrict__ o_beta) {
  __shared__ float Nl[16][16];
  __shared__ float vec[16];
  __shared__ float pm[32], ps[32];
  const int l = threadIdx.x;
  const bool fwd = blockIdx.x < NCH;
  const int c = fwd ? blockIdx.x : (blockIdx.x - NCH);
  const int k0 = c * CHL;
  const int k1 = (k0 + CHL < KMAX) ? k0 + CHL : KMAX;
  if (l < 16) vec[l] = fwd ? ab[(size_t)c * 16 + l] : bb[(size_t)(c + 1) * 16 + l];
  __syncthreads();
  const int steps = k1 - k0;
  for (int si = 0; si < steps; ++si) {
    const int k = fwd ? (k0 + si) : (k1 - 1 - si);
    {   // cooperative 128B-vectorized load of N_k
      const float4* rp = (const float4*)(ltrs + (size_t)k * 256);
      float4 x0 = rp[l * 2], x1 = rp[l * 2 + 1];
      const float4* pp = (const float4*)(lpis + (size_t)(k + 1) * 16);
      float4 p0 = pp[(l & 1) * 2], p1 = pp[(l & 1) * 2 + 1];
      int ii = l >> 1, jb = (l & 1) * 8;
      Nl[ii][jb + 0] = x0.x + p0.x; Nl[ii][jb + 1] = x0.y + p0.y;
      Nl[ii][jb + 2] = x0.z + p0.z; Nl[ii][jb + 3] = x0.w + p0.w;
      Nl[ii][jb + 4] = x1.x + p1.x; Nl[ii][jb + 5] = x1.y + p1.y;
      Nl[ii][jb + 6] = x1.z + p1.z; Nl[ii][jb + 7] = x1.w + p1.w;
    }
    {   // hide L2 latency of the next step matrix
      int kn = fwd ? (k + 1) : (k - 1);
      if (kn >= k0 && kn < k1)
        __builtin_prefetch(ltrs + (size_t)kn * 256 + l * 8, 0, 1);
    }
    __syncthreads();
    float vv[8], mx = -1e30f;
    if (fwd) {      // out[j] = lse_i(vec[i] + N[i][j]); lane pair (j, j+16) splits i
      const int j = l & 15, ib = (l >> 4) * 8;
#pragma unroll
      for (int e = 0; e < 8; ++e) { vv[e] = vec[ib + e] + Nl[ib + e][j]; mx = vv[e] > mx ? vv[e] : mx; }
    } else {        // out[i] = lse_j(N[i][j] + vec[j])
      const int i = l & 15, jb2 = (l >> 4) * 8;
#pragma unroll
      for (int e = 0; e < 8; ++e) { vv[e] = Nl[i][jb2 + e] + vec[jb2 + e]; mx = vv[e] > mx ? vv[e] : mx; }
    }
    float ss = 0.f;
#pragma unroll
    for (int e = 0; e < 8; ++e) ss += expf(vv[e] - mx);
    pm[l] = mx; ps[l] = ss;
    __syncthreads();
    if (l < 16) {
      float m1 = pm[l], s1 = ps[l], m2 = pm[l + 16], s2 = ps[l + 16];
      float M = m1 > m2 ? m1 : m2;
      float o = M + logf(s1 * expf(m1 - M) + s2 * expf(m2 - M));
      vec[l] = o;
      if (fwd) o_alpha[(size_t)(k + 1) * 16 + l] = o;
      else     o_beta[(size_t)k * 16 + l] = o;
    }
    __syncthreads();
  }
}

// ---------------------------------------------------------------------------
extern "C" void kernel_launch(void* const* d_in, const int* in_sizes, int n_in,
                              void* d_out, int out_size, void* d_ws, size_t ws_size,
                              hipStream_t stream) {
  (void)in_sizes; (void)n_in; (void)out_size; (void)ws_size;
  const float* s       = (const float*)d_in[0];
  const float* a_arr   = (const float*)d_in[1];
  const float* pW1     = (const float*)d_in[2];
  const float* pb1     = (const float*)d_in[3];
  const float* pW2     = (const float*)d_in[4];
  const float* pb2     = (const float*)d_in[5];
  const float* pW3     = (const float*)d_in[6];
  const float* pb3     = (const float*)d_in[7];
  const float* oW1     = (const float*)d_in[8];
  const float* ob1     = (const float*)d_in[9];
  const float* oW2     = (const float*)d_in[10];
  const float* ob2     = (const float*)d_in[11];
  const float* oW3     = (const float*)d_in[12];
  const float* ob3     = (const float*)d_in[13];
  const float* alstd   = (const float*)d_in[14];

  // workspace layout (~0.84 MB total)
  _Float16* wp  = (_Float16*)d_ws;
  _Float16* w1p = wp;            // 16384
  _Float16* w2p = wp + 16384;    // 65536
  _Float16* w3p = wp + 81920;    // 32768
  _Float16* w1o = wp + 114688;   // 16384
  _Float16* w2o = wp + 131072;   // 65536
  _Float16* w3o = wp + 196608;   // 69632  -> 266240 halfs
  float* fws    = (float*)((char*)d_ws + (size_t)266240 * sizeof(_Float16));
  float* logtr0 = fws;                     // 16
  float* chunkM = fws + 16;                // 256*256
  float* ab     = chunkM + NCH * 256;      // 257*16
  float* bb     = ab + (NCH + 1) * 16;     // 257*16

  float* out     = (float*)d_out;
  float* o_alpha = out;
  float* o_beta  = out + (size_t)TT * 16;
  float* o_ltrs  = o_beta + (size_t)TT * 16;
  float* o_lpis  = o_ltrs + (size_t)(TT - 1) * 256;
  float* o_ent   = o_lpis + (size_t)TT * 16;

  // Phase 0: pack weights
  pack_w<<<(16384 + 255) / 256, 256, 0, stream>>>(pW1, w1p, 64, 256, 16384);
  pack_w<<<(65536 + 255) / 256, 256, 0, stream>>>(pW2, w2p, 256, 256, 65536);
  pack_w<<<(32768 + 255) / 256, 256, 0, stream>>>(pW3, w3p, 256, 128, 32768);
  pack_w<<<(16384 + 255) / 256, 256, 0, stream>>>(oW1, w1o, 64, 256, 16384);
  pack_w<<<(65536 + 255) / 256, 256, 0, stream>>>(oW2, w2o, 256, 256, 65536);
  pack_w<<<(69632 + 255) / 256, 256, 0, stream>>>(oW3, w3o, 256, 272, 69632);

  // Phase 1: fused dual-MLP over T rows (WMMA + TDM)
  mlp_kernel<<<TT / 16, 32, 0, stream>>>(s, a_arr, w1p, w2p, w3p, w1o, w2o, w3o,
                                         pb1, pb2, pb3, ob1, ob2, ob3, alstd,
                                         o_lpis, o_ltrs, o_ent, logtr0);

  // Phase 2: chunked log-semiring scans
  scan_chunks<<<NCH, 256, 0, stream>>>(o_ltrs, o_lpis, chunkM);
  scan_bounds<<<1, 32, 0, stream>>>(chunkM, logtr0, o_lpis, ab, bb, o_alpha, o_beta);
  scan_fill<<<2 * NCH, 32, 0, stream>>>(o_ltrs, o_lpis, ab, bb, o_alpha, o_beta);
}
